// DynamicPerceiverEncoder_70884140253285
// MI455X (gfx1250) — matrix operand order repaired and checked by reading, not compiled, not run on hardware
//
#include <hip/hip_runtime.h>
#include <math.h>

typedef __attribute__((ext_vector_type(16))) __bf16 v16bf;
typedef __attribute__((ext_vector_type(8)))  __bf16 v8bf;
typedef __attribute__((ext_vector_type(8)))  float  v8f;

#define DMODEL 768
#define NBATCH 4
#define NTOK   196
#define NVIS   49
#define NMASK  147
#define NLAT   256
#define NHEAD  8
#define DHEAD  96
#define NLAYER 6
#define NEXP   8
#define DFF    1536
#define DQKV   2304
#define KPAD_X 64          // padded kv length for cross attention (49 -> 64)
#define ATTN_SCALE 0.1020620726159658f   // 1/sqrt(96)

// ---------------------------------------------------------------------------
// helpers
// ---------------------------------------------------------------------------
__device__ __forceinline__ float gelu_exact(float x) {
  return 0.5f * x * (1.0f + erff(x * 0.7071067811865475f));
}

// sincos 2D position embedding, matching the numpy reference for D=768, 14x14
__device__ __forceinline__ float pos_embed(int t, int d) {
  int p  = (d < 384) ? (t % 14) : (t / 14);   // first half uses w, second uses h
  int dd = (d < 384) ? d : (d - 384);
  int m  = (dd < 192) ? dd : (dd - 192);
  float omega = expf(-(float)m * (9.210340371976184f / 192.0f)); // 10000^(-m/192)
  float v = (float)p * omega;
  return (dd < 192) ? sinf(v) : cosf(v);
}

// A fragment (16x32 bf16): elements 0..15 -> k = 16g+e, contiguous
__device__ __forceinline__ v16bf load_a_frag(const float* __restrict__ arow,
                                             int kb, int g) {
  v16bf af;
  const float* ap0 = arow + kb + 16 * g;
  const float* ap1 = arow + kb + 16 * g + 8;
#pragma unroll
  for (int e = 0; e < 8; ++e) { af[e] = (__bf16)ap0[e]; af[e + 8] = (__bf16)ap1[e]; }
  return af;
}

// B fragment (32x16 bf16) from (N,K) row-major weights: k = 16g+e, contiguous
__device__ __forceinline__ v16bf load_b_frag(const float* __restrict__ brow,
                                             int kb, int g) {
  v16bf bf;
  const float* bp = brow + kb + 16 * g;
#pragma unroll
  for (int e = 0; e < 16; ++e) bf[e] = (__bf16)bp[e];
  return bf;
}

// B fragment from an LDS row: 32 bytes at a 16B-aligned offset -> 2x ds_load_b128
__device__ __forceinline__ v16bf lds_b_frag(const __bf16* row, int g) {
  const v8bf* p = (const v8bf*)(row + 16 * g);
  v8bf lo = p[0], hi = p[1];
  return __builtin_shufflevector(lo, hi, 0, 1, 2, 3, 4, 5, 6, 7,
                                 8, 9, 10, 11, 12, 13, 14, 15);
}

// ---------------------------------------------------------------------------
// Generic batched WMMA GEMM:  Out[m,n] = act( sum_k A[m,k] * W[n,k] + bias[n] )
// W is (N,K) row-major.  Block = 128(M) x 32(N): the 32x32 K-step B tile is
// cooperatively staged once per block into LDS (converted to bf16 once), then
// the 4 waves each compute a 32(M) x 32(N) register tile (2x2 WMMA tiles,
// A fragments reused across N, B fragments reused across M).
// Batch (grid.z) offset decomposes as z = zo*zdiv + zi with separate strides,
// so all (batch, head) attention GEMMs fold into one launch.
// Optional: residual add, exact GELU, scaled accumulation Out += rowScale[m]*v.
// K must be a multiple of 32.
// ---------------------------------------------------------------------------
__global__ void wmma_gemm(
    const float* __restrict__ A, int sA, long long bAo, long long bAi,
    const float* __restrict__ W, int sW, long long bWo, long long bWi,
    const float* __restrict__ bias,
    const float* __restrict__ res, int sRes,
    const float* __restrict__ rowScale, int rsStride,
    float* __restrict__ Out, int sOut, long long bOo, long long bOi,
    int zdiv, int M, int N, int K, int act, int accum)
{
  // 32 n-rows x 32 k of bf16; row stride 40 (80B) => 16B-aligned rows that
  // land on 16 disjoint 4-bank groups -> conflict-free 16B ds loads.
  __shared__ __bf16 Bs[32][40];

  const int lane  = threadIdx.x;            // 0..31 (wave32)
  const int wy    = threadIdx.y;            // wave 0..3
  const int tid   = wy * 32 + lane;         // 0..127
  const int tileN = blockIdx.x * 32;
  const int tileM = blockIdx.y * 128 + wy * 32;

  const int z  = blockIdx.z;
  const int zo = z / zdiv, zi = z - zo * zdiv;
  A   += zo * bAo + zi * bAi;
  W   += zo * bWo + zi * bWi;
  Out += zo * bOo + zi * bOi;

  const int g = lane >> 4;                  // lane half
  int mrow0 = tileM + (lane & 15);      if (mrow0 >= M) mrow0 = M - 1;
  int mrow1 = tileM + 16 + (lane & 15); if (mrow1 >= M) mrow1 = M - 1;
  const float* arow0 = A + (long long)mrow0 * sA;
  const float* arow1 = A + (long long)mrow1 * sA;

  // cooperative staging assignment: 128 threads cover 32 rows x 4 k-chunks
  const int nl   = tid >> 2;                // 0..31
  const int koff = (tid & 3) * 8;           // 0,8,16,24
  int ncl = tileN + nl; if (ncl >= N) ncl = N - 1;
  const float* wrow = W + (long long)ncl * sW + koff;

  v8f acc00 = {}, acc01 = {}, acc10 = {}, acc11 = {};
  for (int kb = 0; kb < K; kb += 32) {
    __syncthreads();                        // previous K-step consumed
    __builtin_prefetch(wrow + kb + 32, 0, 3);       // global_prefetch_b8
    {
      v8bf st;
#pragma unroll
      for (int i = 0; i < 8; ++i) st[i] = (__bf16)wrow[kb + i];
      *(v8bf*)&Bs[nl][koff] = st;           // one ds_store_b128
    }
    __syncthreads();                        // tile staged

    v16bf a0 = load_a_frag(arow0, kb, g);
    v16bf a1 = load_a_frag(arow1, kb, g);
    v16bf b0 = lds_b_frag(&Bs[lane & 15][0], g);
    v16bf b1 = lds_b_frag(&Bs[16 + (lane & 15)][0], g);

    acc00 = __builtin_amdgcn_wmma_f32_16x16x32_bf16(false, a0, false, b0,
                                                    (short)0, acc00, false, false);
    acc01 = __builtin_amdgcn_wmma_f32_16x16x32_bf16(false, a0, false, b1,
                                                    (short)0, acc01, false, false);
    acc10 = __builtin_amdgcn_wmma_f32_16x16x32_bf16(false, a1, false, b0,
                                                    (short)0, acc10, false, false);
    acc11 = __builtin_amdgcn_wmma_f32_16x16x32_bf16(false, a1, false, b1,
                                                    (short)0, acc11, false, false);
  }

  v8f accs[2][2] = {{acc00, acc01}, {acc10, acc11}};
#pragma unroll
  for (int mt = 0; mt < 2; ++mt) {
#pragma unroll
    for (int nt = 0; nt < 2; ++nt) {
      const int n = tileN + 16 * nt + (lane & 15);
      if (n < N) {
        const float bv = bias ? bias[n] : 0.0f;
#pragma unroll
        for (int r = 0; r < 8; ++r) {
          int m = tileM + 16 * mt + r + 8 * g;
          if (m < M) {
            float v = accs[mt][nt][r] + bv;
            if (act == 1) v = gelu_exact(v);
            if (res) v += res[(long long)m * sRes + n];
            float* o = Out + (long long)m * sOut + n;
            if (accum) {
              float s = rowScale ? rowScale[(long long)m * rsStride] : 1.0f;
              *o += s * v;
            } else {
              *o = v;
            }
          }
        }
      }
    }
  }
}

// ---------------------------------------------------------------------------
// Patch conv (16x16 stride 16) as implicit-im2col WMMA GEMM, fused with
// conv bias + sincos pos-embed + type embedding.  M = B*196, N = K = 768.
// ---------------------------------------------------------------------------
__global__ void conv_patch_kernel(const float* __restrict__ img,
                                  const float* __restrict__ cw,
                                  const float* __restrict__ cb,
                                  const int*   __restrict__ type_ids,
                                  const float* __restrict__ type_emb,
                                  float* __restrict__ patches)
{
  const int M = NBATCH * NTOK, K = 768;
  const int lane  = threadIdx.x;
  const int tileN = blockIdx.x * 32;
  const int tileM = (blockIdx.y * 4 + threadIdx.y) * 16;
  if (tileM >= M) return;

  const int g = lane >> 4;
  int mrow = tileM + (lane & 15); if (mrow >= M) mrow = M - 1;
  const int b  = mrow / NTOK, t = mrow % NTOK;
  const int ph = t / 14,      pw = t % 14;
  const int ncol0 = tileN + (lane & 15);        // N = 768 exactly tiled
  const int ncol1 = tileN + 16 + (lane & 15);
  const float* brow0 = cw + (long long)ncol0 * K;
  const float* brow1 = cw + (long long)ncol1 * K;

  v8f acc0 = {}, acc1 = {};
  for (int kb = 0; kb < K; kb += 32) {
    v16bf af;
    // im2col: k -> (c = k/256, i = (k%256)/16, j = k%16); 8-aligned chunks
    // never cross a row of the 16x16 patch, so each chunk is 8 contiguous px.
    {
      int k0 = kb + 16 * g;
      int c = k0 >> 8, rem = k0 & 255, i = rem >> 4, j = rem & 15;
      const float* p = img + ((long long)(b * 3 + c) * 224 + (16 * ph + i)) * 224
                           + 16 * pw + j;
#pragma unroll
      for (int e = 0; e < 8; ++e) af[e] = (__bf16)p[e];
    }
    {
      int k1 = kb + 16 * g + 8;
      int c = k1 >> 8, rem = k1 & 255, i = rem >> 4, j = rem & 15;
      const float* p = img + ((long long)(b * 3 + c) * 224 + (16 * ph + i)) * 224
                           + 16 * pw + j;
#pragma unroll
      for (int e = 0; e < 8; ++e) af[e + 8] = (__bf16)p[e];
    }
    v16bf bf0 = load_b_frag(brow0, kb, g);
    v16bf bf1 = load_b_frag(brow1, kb, g);
    acc0 = __builtin_amdgcn_wmma_f32_16x16x32_bf16(false, af, false, bf0,
                                                   (short)0, acc0, false, false);
    acc1 = __builtin_amdgcn_wmma_f32_16x16x32_bf16(false, af, false, bf1,
                                                   (short)0, acc1, false, false);
  }

  v8f accs[2] = {acc0, acc1};
#pragma unroll
  for (int j = 0; j < 2; ++j) {
    const int n = tileN + 16 * j + (lane & 15);
    const float bv = cb[n];
#pragma unroll
    for (int r = 0; r < 8; ++r) {
      int m = tileM + r + 8 * g;
      if (m < M) {
        int b2 = m / NTOK, t2 = m % NTOK;
        float v = accs[j][r] + bv + pos_embed(t2, n)
                + type_emb[(long long)type_ids[b2] * DMODEL + n];
        patches[(long long)m * DMODEL + n] = v;
      }
    }
  }
}

// ---------------------------------------------------------------------------
// V transposes: produce (b,h,d,k) layout so att@V is a contiguous-B GEMM.
// ---------------------------------------------------------------------------
__global__ void transpose_v_self_kernel(const float* __restrict__ qkv,
                                        float* __restrict__ vT)
{
  long long i = (long long)blockIdx.x * blockDim.x + threadIdx.x;
  const long long total = (long long)NBATCH * NHEAD * DHEAD * NLAT;
  if (i >= total) return;
  int t = (int)(i % NLAT); long long r = i / NLAT;
  int d = (int)(r % DHEAD); r /= DHEAD;
  int h = (int)(r % NHEAD); int b = (int)(r / NHEAD);
  vT[i] = qkv[((long long)b * NLAT + t) * DQKV + 2 * DMODEL + h * DHEAD + d];
}

__global__ void transpose_v_cross_kernel(const float* __restrict__ kv,
                                         float* __restrict__ vT)
{
  long long i = (long long)blockIdx.x * blockDim.x + threadIdx.x;
  const long long total = (long long)NBATCH * NHEAD * DHEAD * KPAD_X;
  if (i >= total) return;
  int k = (int)(i % KPAD_X); long long r = i / KPAD_X;
  int d = (int)(r % DHEAD); r /= DHEAD;
  int h = (int)(r % NHEAD); int b = (int)(r / NHEAD);
  vT[i] = (k < NVIS)
        ? kv[((long long)b * NVIS + k) * (2 * DMODEL) + DMODEL + h * DHEAD + d]
        : 0.0f;
}

// ---------------------------------------------------------------------------
// Stable argsort of the 196 noise values per batch (rank counting == stable)
// ---------------------------------------------------------------------------
__global__ void argsort_kernel(const float* __restrict__ noise, int* __restrict__ ids)
{
  const int b = blockIdx.x, i = threadIdx.x;
  if (i >= NTOK) return;
  const float xi = noise[b * NTOK + i];
  int rank = 0;
  for (int j = 0; j < NTOK; ++j) {
    float xj = noise[b * NTOK + j];
    rank += (xj < xi) || (xj == xi && j < i);
  }
  ids[b * NTOK + rank] = i;
}

__global__ void write_ids_kernel(const int* __restrict__ ids, float* __restrict__ out)
{
  const int b = blockIdx.x, i = threadIdx.x;
  if (i < NVIS)       out[b * NVIS + i] = (float)ids[b * NTOK + i];
  else if (i < NTOK)  out[NBATCH * NVIS + b * NMASK + (i - NVIS)]
                        = (float)ids[b * NTOK + i];
}

__global__ void gather_kernel(const float* __restrict__ patches,
                              const int* __restrict__ ids,
                              float* __restrict__ vis)
{
  const int bv = blockIdx.x;               // 0..B*NVIS-1
  const int b = bv / NVIS, v = bv % NVIS;
  const int src = ids[b * NTOK + v];
  const float* p = patches + ((long long)b * NTOK + src) * DMODEL;
  float* o = vis + (long long)bv * DMODEL;
  for (int d = threadIdx.x; d < DMODEL; d += blockDim.x) o[d] = p[d];
}

__global__ void bcast_lat_kernel(const float* __restrict__ latents, float* __restrict__ lat)
{
  long long i = (long long)blockIdx.x * blockDim.x + threadIdx.x;
  if (i < (long long)NBATCH * NLAT * DMODEL) lat[i] = latents[i % (NLAT * DMODEL)];
}

// ---------------------------------------------------------------------------
// Row softmax over attention scores; scales by 1/sqrt(dh) and zeroes the pad
// ---------------------------------------------------------------------------
__global__ void softmax_kernel(float* __restrict__ s, int Klen, int Kpad, float scale)
{
  __shared__ float sm[256];
  const long long row = blockIdx.x;
  float* sr = s + row * (long long)Kpad;
  const int tid = threadIdx.x;

  float mx = -3.4e38f;
  for (int i = tid; i < Klen; i += 256) mx = fmaxf(mx, sr[i] * scale);
  sm[tid] = mx; __syncthreads();
  for (int off = 128; off > 0; off >>= 1) {
    if (tid < off) sm[tid] = fmaxf(sm[tid], sm[tid + off]);
    __syncthreads();
  }
  const float m = sm[0]; __syncthreads();

  float sum = 0.0f;
  for (int i = tid; i < Klen; i += 256) sum += expf(sr[i] * scale - m);
  sm[tid] = sum; __syncthreads();
  for (int off = 128; off > 0; off >>= 1) {
    if (tid < off) sm[tid] += sm[tid + off];
    __syncthreads();
  }
  const float inv = 1.0f / sm[0];

  for (int i = tid; i < Kpad; i += 256)
    sr[i] = (i < Klen) ? expf(sr[i] * scale - m) * inv : 0.0f;
}

// ---------------------------------------------------------------------------
// LayerNorm over last dim (768)
// ---------------------------------------------------------------------------
__global__ void ln_kernel(const float* __restrict__ x, const float* __restrict__ w,
                          const float* __restrict__ b, float* __restrict__ out)
{
  __shared__ float sm[256];
  const long long row = blockIdx.x;
  const float* xr = x + row * DMODEL;
  float* orow = out + row * DMODEL;
  const int tid = threadIdx.x;

  float s = 0.0f;
  for (int i = tid; i < DMODEL; i += 256) s += xr[i];
  sm[tid] = s; __syncthreads();
  for (int off = 128; off > 0; off >>= 1) {
    if (tid < off) sm[tid] += sm[tid + off];
    __syncthreads();
  }
  const float mean = sm[0] * (1.0f / DMODEL); __syncthreads();

  float v = 0.0f;
  for (int i = tid; i < DMODEL; i += 256) { float d = xr[i] - mean; v += d * d; }
  sm[tid] = v; __syncthreads();
  for (int off = 128; off > 0; off >>= 1) {
    if (tid < off) sm[tid] += sm[tid + off];
    __syncthreads();
  }
  const float rstd = rsqrtf(sm[0] * (1.0f / DMODEL) + 1e-5f);

  for (int i = tid; i < DMODEL; i += 256)
    orow[i] = (xr[i] - mean) * rstd * w[i] + b[i];
}

// ---------------------------------------------------------------------------
// Router: softmax over 8 experts, top-2, softmax of the two probs -> dense comb
// ---------------------------------------------------------------------------
__global__ void router_top2_kernel(const float* __restrict__ logits,
                                   float* __restrict__ comb, int T)
{
  const int t = blockIdx.x * blockDim.x + threadIdx.x;
  if (t >= T) return;
  const float* lr = logits + (long long)t * NEXP;
  float p[NEXP];
  float mx = lr[0];
  for (int e = 1; e < NEXP; ++e) mx = fmaxf(mx, lr[e]);
  float sum = 0.0f;
  for (int e = 0; e < NEXP; ++e) { p[e] = expf(lr[e] - mx); sum += p[e]; }
  const float inv = 1.0f / sum;
  for (int e = 0; e < NEXP; ++e) p[e] *= inv;

  int i0 = 0; float v0 = p[0];
  for (int e = 1; e < NEXP; ++e) if (p[e] > v0) { v0 = p[e]; i0 = e; }
  int i1 = -1; float v1 = -1.0f;
  for (int e = 0; e < NEXP; ++e) if (e != i0 && p[e] > v1) { v1 = p[e]; i1 = e; }

  const float e1 = expf(v1 - v0);
  const float w0 = 1.0f / (1.0f + e1), w1 = e1 / (1.0f + e1);
  for (int e = 0; e < NEXP; ++e) comb[(long long)t * NEXP + e] = 0.0f;
  comb[(long long)t * NEXP + i0] = w0;
  comb[(long long)t * NEXP + i1] = w1;
}

// ---------------------------------------------------------------------------
// Host-side launcher helper
// ---------------------------------------------------------------------------
static inline void launch_gemm(hipStream_t st,
    const float* A, int sA, long long bAo, long long bAi,
    const float* W, int sW, long long bWo, long long bWi,
    const float* bias, const float* res, int sRes,
    const float* rs, int rsStride,
    float* Out, int sOut, long long bOo, long long bOi,
    int zdiv, int M, int N, int K, int act, int accum, int Z)
{
  dim3 blk(32, 4);
  dim3 grd((N + 31) / 32, (M + 127) / 128, Z);
  wmma_gemm<<<grd, blk, 0, st>>>(A, sA, bAo, bAi, W, sW, bWo, bWi,
                                 bias, res, sRes, rs, rsStride,
                                 Out, sOut, bOo, bOi,
                                 zdiv, M, N, K, act, accum);
}

extern "C" void kernel_launch(void* const* d_in, const int* in_sizes, int n_in,
                              void* d_out, int out_size, void* d_ws, size_t ws_size,
                              hipStream_t stream)
{
  (void)in_sizes; (void)n_in; (void)out_size; (void)ws_size;

  const float* images    = (const float*)d_in[0];
  const int*   type_ids  = (const int*)  d_in[1];
  const float* noise     = (const float*)d_in[2];
  const float* conv_w    = (const float*)d_in[3];
  const float* conv_b    = (const float*)d_in[4];
  const float* latents   = (const float*)d_in[5];
  const float* type_emb  = (const float*)d_in[6];
  const float* ca_in_w   = (const float*)d_in[7];
  const float* ca_in_b   = (const float*)d_in[8];
  const float* ca_out_w  = (const float*)d_in[9];
  const float* ca_out_b  = (const float*)d_in[10];
  const float* ca_ln_w   = (const float*)d_in[11];
  const float* ca_ln_b   = (const float*)d_in[12];
  const float* blk_in_w  = (const float*)d_in[13];
  const float* blk_in_b  = (const float*)d_in[14];
  const float* blk_out_w = (const float*)d_in[15];
  const float* blk_out_b = (const float*)d_in[16];
  const float* blk_ln1_w = (const float*)d_in[17];
  const float* blk_ln1_b = (const float*)d_in[18];
  const float* blk_ln2_w = (const float*)d_in[19];
  const float* blk_ln2_b = (const float*)d_in[20];
  const float* router_w  = (const float*)d_in[21];
  const float* router_b  = (const float*)d_in[22];
  const float* e_w1      = (const float*)d_in[23];
  const float* e_b1      = (const float*)d_in[24];
  const float* e_w2      = (const float*)d_in[25];
  const float* e_b2      = (const float*)d_in[26];

  // ---- workspace carving (floats) ----
  float* ws      = (float*)d_ws;
  float* patches = ws; ws += (long long)NBATCH * NTOK * DMODEL;
  float* visible = ws; ws += (long long)NBATCH * NVIS * DMODEL;
  float* qbuf    = ws; ws += (long long)NLAT * DMODEL;
  float* kvbuf   = ws; ws += (long long)NBATCH * NVIS * 2 * DMODEL;
  float* scores  = ws; ws += (long long)NBATCH * NHEAD * NLAT * NLAT;
  float* obuf    = ws; ws += (long long)NBATCH * NLAT * DMODEL;
  float* qkvbuf  = ws; ws += (long long)NBATCH * NLAT * DQKV;
  float* vTs     = ws; ws += (long long)NBATCH * NHEAD * DHEAD * NLAT;
  float* vTx     = ws; ws += (long long)NBATCH * NHEAD * DHEAD * KPAD_X;
  float* lat0    = ws; ws += (long long)NBATCH * NLAT * DMODEL;
  float* t0      = ws; ws += (long long)NBATCH * NLAT * DMODEL;
  float* t1      = ws; ws += (long long)NBATCH * NLAT * DMODEL;
  float* t2      = ws; ws += (long long)NBATCH * NLAT * DMODEL;
  float* hbuf    = ws; ws += (long long)NBATCH * NLAT * DFF;
  float* logits  = ws; ws += (long long)NBATCH * NLAT * NEXP;
  float* comb    = ws; ws += (long long)NBATCH * NLAT * NEXP;
  int*   ids     = (int*)ws;                 // B*196 ints

  const int Mtok = NBATCH * NLAT;            // 1024 latent rows total

  // ---- patch embed: conv-as-GEMM fused with pos/type embed ----
  conv_patch_kernel<<<dim3(DMODEL / 32, 13), dim3(32, 4), 0, stream>>>(
      images, conv_w, conv_b, type_ids, type_emb, patches);

  // ---- masking ----
  argsort_kernel<<<NBATCH, 256, 0, stream>>>(noise, ids);
  write_ids_kernel<<<NBATCH, 256, 0, stream>>>(
      ids, (float*)d_out + (long long)NBATCH * NLAT * DMODEL);
  gather_kernel<<<NBATCH * NVIS, 256, 0, stream>>>(patches, ids, visible);

  // ---- cross attention ----
  // q = latents @ wq.T + bq  (shared across batch since lat is broadcast)
  launch_gemm(stream, latents, DMODEL, 0, 0,
              ca_in_w, DMODEL, 0, 0, ca_in_b,
              nullptr, 0, nullptr, 0,
              qbuf, DMODEL, 0, 0, 1, NLAT, DMODEL, DMODEL, 0, 0, 1);
  // kv = visible @ [wk;wv].T + [bk;bv]
  launch_gemm(stream, visible, DMODEL, 0, 0,
              ca_in_w + (long long)DMODEL * DMODEL, DMODEL, 0, 0,
              ca_in_b + DMODEL, nullptr, 0, nullptr, 0,
              kvbuf, 2 * DMODEL, 0, 0, 1,
              NBATCH * NVIS, 2 * DMODEL, DMODEL, 0, 0, 1);
  transpose_v_cross_kernel<<<
      (int)(((long long)NBATCH * NHEAD * DHEAD * KPAD_X + 255) / 256), 256, 0,
      stream>>>(kvbuf, vTx);
  // scores[b,h] = q @ k.T   (z = b*8 + h, one launch)
  launch_gemm(stream, qbuf, DMODEL, 0, DHEAD,
              kvbuf, 2 * DMODEL, (long long)NVIS * 2 * DMODEL, DHEAD,
              nullptr, nullptr, 0, nullptr, 0,
              scores, KPAD_X, (long long)NHEAD * NLAT * KPAD_X,
              (long long)NLAT * KPAD_X,
              NHEAD, NLAT, NVIS, DHEAD, 0, 0, NBATCH * NHEAD);
  softmax_kernel<<<NBATCH * NHEAD * NLAT, 256, 0, stream>>>(
      scores, NVIS, KPAD_X, ATTN_SCALE);
  // o[b,h] = att @ v  via vT (contiguous B fragments)
  launch_gemm(stream, scores, KPAD_X, (long long)NHEAD * NLAT * KPAD_X,
              (long long)NLAT * KPAD_X,
              vTx, KPAD_X, (long long)NHEAD * DHEAD * KPAD_X,
              (long long)DHEAD * KPAD_X,
              nullptr, nullptr, 0, nullptr, 0,
              obuf, DMODEL, (long long)NLAT * DMODEL, DHEAD,
              NHEAD, NLAT, DHEAD, KPAD_X, 0, 0, NBATCH * NHEAD);
  bcast_lat_kernel<<<(NBATCH * NLAT * DMODEL + 255) / 256, 256, 0, stream>>>(
      latents, lat0);
  launch_gemm(stream, obuf, DMODEL, 0, 0,
              ca_out_w, DMODEL, 0, 0, ca_out_b,
              lat0, DMODEL, nullptr, 0,
              t0, DMODEL, 0, 0, 1, Mtok, DMODEL, DMODEL, 0, 0, 1);
  ln_kernel<<<Mtok, 256, 0, stream>>>(t0, ca_ln_w, ca_ln_b, lat0);

  // ---- transformer blocks ----
  for (int l = 0; l < NLAYER; ++l) {
    // qkv projection
    launch_gemm(stream, lat0, DMODEL, 0, 0,
                blk_in_w + (long long)l * DQKV * DMODEL, DMODEL, 0, 0,
                blk_in_b + (long long)l * DQKV, nullptr, 0, nullptr, 0,
                qkvbuf, DQKV, 0, 0, 1, Mtok, DQKV, DMODEL, 0, 0, 1);
    transpose_v_self_kernel<<<
        (int)(((long long)NBATCH * NHEAD * DHEAD * NLAT + 255) / 256), 256, 0,
        stream>>>(qkvbuf, vTs);
    // scores[b,h] = q @ k.T
    launch_gemm(stream, qkvbuf, DQKV, (long long)NLAT * DQKV, DHEAD,
                qkvbuf + DMODEL, DQKV, (long long)NLAT * DQKV, DHEAD,
                nullptr, nullptr, 0, nullptr, 0,
                scores, NLAT, (long long)NHEAD * NLAT * NLAT,
                (long long)NLAT * NLAT,
                NHEAD, NLAT, NLAT, DHEAD, 0, 0, NBATCH * NHEAD);
    softmax_kernel<<<NBATCH * NHEAD * NLAT, 256, 0, stream>>>(
        scores, NLAT, NLAT, ATTN_SCALE);
    // o[b,h] = att @ v
    launch_gemm(stream, scores, NLAT, (long long)NHEAD * NLAT * NLAT,
                (long long)NLAT * NLAT,
                vTs, NLAT, (long long)NHEAD * DHEAD * NLAT,
                (long long)DHEAD * NLAT,
                nullptr, nullptr, 0, nullptr, 0,
                obuf, DMODEL, (long long)NLAT * DMODEL, DHEAD,
                NHEAD, NLAT, DHEAD, NLAT, 0, 0, NBATCH * NHEAD);
    // out projection + residual
    launch_gemm(stream, obuf, DMODEL, 0, 0,
                blk_out_w + (long long)l * DMODEL * DMODEL, DMODEL, 0, 0,
                blk_out_b + (long long)l * DMODEL, lat0, DMODEL, nullptr, 0,
                t0, DMODEL, 0, 0, 1, Mtok, DMODEL, DMODEL, 0, 0, 1);
    ln_kernel<<<Mtok, 256, 0, stream>>>(
        t0, blk_ln1_w + l * DMODEL, blk_ln1_b + l * DMODEL, t1);

    // MoE router
    launch_gemm(stream, t1, DMODEL, 0, 0,
                router_w + (long long)l * NEXP * DMODEL, DMODEL, 0, 0,
                router_b + (long long)l * NEXP, nullptr, 0, nullptr, 0,
                logits, NEXP, 0, 0, 1, Mtok, NEXP, DMODEL, 0, 0, 1);
    router_top2_kernel<<<(Mtok + 255) / 256, 256, 0, stream>>>(logits, comb, Mtok);

    // y = x1 (residual), then accumulate comb[:,e] * expert_e(x1)
    hipMemcpyAsync(t2, t1, sizeof(float) * (size_t)Mtok * DMODEL,
                   hipMemcpyDeviceToDevice, stream);
    for (int e = 0; e < NEXP; ++e) {
      const float* w1 = e_w1 + ((long long)l * NEXP + e) * DFF * DMODEL;
      const float* b1 = e_b1 + ((long long)l * NEXP + e) * DFF;
      const float* w2 = e_w2 + ((long long)l * NEXP + e) * DMODEL * DFF;
      const float* b2 = e_b2 + ((long long)l * NEXP + e) * DMODEL;
      launch_gemm(stream, t1, DMODEL, 0, 0, w1, DMODEL, 0, 0, b1,
                  nullptr, 0, nullptr, 0,
                  hbuf, DFF, 0, 0, 1, Mtok, DFF, DMODEL, 1 /*gelu*/, 0, 1);
      launch_gemm(stream, hbuf, DFF, 0, 0, w2, DFF, 0, 0, b2,
                  nullptr, 0, comb + e, NEXP,
                  t2, DMODEL, 0, 0, 1, Mtok, DMODEL, DFF, 0, 1 /*accum*/, 1);
    }
    ln_kernel<<<Mtok, 256, 0, stream>>>(
        t2, blk_ln2_w + l * DMODEL, blk_ln2_b + l * DMODEL, lat0);
  }

  // ---- final output copy ----
  hipMemcpyAsync(d_out, lat0, sizeof(float) * (size_t)Mtok * DMODEL,
                 hipMemcpyDeviceToDevice, stream);
}
